// DVAE_73847667687625
// MI455X (gfx1250) — matrix operand order repaired and checked by reading, not compiled, not run on hardware
//
#include <hip/hip_runtime.h>
#include <math.h>

#define DEV_INLINE __device__ __forceinline__

typedef __attribute__((ext_vector_type(16))) __bf16 v16bf;
typedef __attribute__((ext_vector_type(8)))  float  v8f;

namespace {
enum : int {
  B_   = 2048,
  MAXN = 16,
  NVT_ = 8,
  HS_  = 501,
  NZ_  = 56,
  VS_  = HS_ + MAXN,   // 517
  HSP  = 512,          // HS padded to mult of 32 (GEMM K)
  G3   = 3 * HS_,      // 1503
  G3P  = 1536,         // padded N for w_hh GEMM
  GMN  = 1024          // padded N for [gate;map] GEMM (gate rows 0..500, map rows 512..1012)
};
}

DEV_INLINE unsigned short f2bf(float f) {
  unsigned int u = __float_as_uint(f);
  u += 0x7FFFu + ((u >> 16) & 1u);   // round-to-nearest-even
  return (unsigned short)(u >> 16);
}
DEV_INLINE float sigmoidf_(float x) { return 1.0f / (1.0f + __expf(-x)); }

union FragBF { v16bf v; unsigned int u[8]; };

// ---------------------------------------------------------------------------
// C (M x ldc, f32) = A (M x 512, bf16) * W^T, W stored N x 512 bf16 row-major.
// Block: 256 threads = 8 waves, 2(M) x 4(N); wave tile 32x64 via 8 accumulators.
// K-loop fully unrolled with double-buffered fragments so loads for step s+1
// overlap the 8 WMMAs of step s.
// ---------------------------------------------------------------------------
__global__ __launch_bounds__(256) void wmma_gemm_bf16(
    const unsigned short* __restrict__ A,
    const unsigned short* __restrict__ W,
    float* __restrict__ C, int ldc)
{
  const int lane = threadIdx.x & 31;
  const int wave = threadIdx.x >> 5;
  const int wm   = wave & 1;                       // 2 M-waves
  const int wn   = wave >> 1;                      // 4 N-waves
  const int row0 = blockIdx.y * 64 + wm * 32;
  const int col0 = blockIdx.x * 256 + wn * 64;
  const int mr   = lane & 15;
  const int half = lane >> 4;

  const unsigned short* aRow0 = A + (size_t)(row0 + mr) * HSP;
  const unsigned short* aRow1 = aRow0 + (size_t)16 * HSP;
  const unsigned short* w0    = W + (size_t)(col0 + mr) * HSP;
  const unsigned short* w1    = w0 + (size_t)16 * HSP;
  const unsigned short* w2    = w0 + (size_t)32 * HSP;
  const unsigned short* w3    = w0 + (size_t)48 * HSP;

  __builtin_prefetch(w0, 0, 1);
  __builtin_prefetch(w1, 0, 1);
  __builtin_prefetch(w2, 0, 1);
  __builtin_prefetch(w3, 0, 1);

  v8f acc[2][4];
#pragma unroll
  for (int i = 0; i < 2; ++i)
#pragma unroll
    for (int n = 0; n < 4; ++n)
      acc[i][n] = (v8f){};

  FragBF fa[2][2];   // [buf][m-tile]
  FragBF fb[2][4];   // [buf][n-tile]

  auto ld32 = [](const unsigned short* p) -> unsigned int {
    return *(const unsigned int*)p;
  };

  auto loadSet = [&](int buf, int k0) {
#pragma unroll
    for (int j = 0; j < 8; ++j) {
      // A 16x32 bf16 layout (ISA 7.12.2): lane = row; VGPR j holds K pair
      //   j<4 : K = 2j + 8*half ;  j>=4 : K = 16 + 2(j-4) + 8*half
      const int ka = k0 + ((j >> 2) << 4) + (half << 3) + ((j & 3) << 1);
      // B 32x16: lane = column; same K mapping as A
      const int kb = k0 + ((j >> 2) << 4) + (half << 3) + ((j & 3) << 1);
      fa[buf][0].u[j] = ld32(aRow0 + ka);
      fa[buf][1].u[j] = ld32(aRow1 + ka);
      fb[buf][0].u[j] = ld32(w0 + kb);
      fb[buf][1].u[j] = ld32(w1 + kb);
      fb[buf][2].u[j] = ld32(w2 + kb);
      fb[buf][3].u[j] = ld32(w3 + kb);
    }
  };

  loadSet(0, 0);
#pragma unroll
  for (int s = 0; s < HSP / 32; ++s) {             // 16 steps, fully unrolled
    const int cur = s & 1;
    if (s < HSP / 32 - 1) loadSet(cur ^ 1, (s + 1) * 32);
#pragma unroll
    for (int i = 0; i < 2; ++i)
#pragma unroll
      for (int n = 0; n < 4; ++n)
        acc[i][n] = __builtin_amdgcn_wmma_f32_16x16x32_bf16(
            false, fa[cur][i].v, false, fb[cur][n].v,
            (short)0, acc[i][n], false, false);
  }

  // C/D layout: lane = column (lane&15); VGPR r -> row r (lanes 0-15), r+8 (16-31)
#pragma unroll
  for (int i = 0; i < 2; ++i) {
    float* cp = C + (size_t)(row0 + 16 * i + 8 * half) * ldc + col0 + mr;
#pragma unroll
    for (int r = 0; r < 8; ++r)
#pragma unroll
      for (int n = 0; n < 4; ++n)
        cp[(size_t)r * ldc + 16 * n] = acc[i][n][r];
  }
}

// ---------------------------------------------------------------------------
// Preprocess: fp32 weights -> padded bf16
// ---------------------------------------------------------------------------
__global__ __launch_bounds__(256) void prep_whh(const float* __restrict__ w_hh,
                                                unsigned short* __restrict__ whh_b)
{
  int idx = blockIdx.x * 256 + threadIdx.x;       // G3P * HSP
  if (idx >= G3P * HSP) return;
  int r = idx >> 9, k = idx & (HSP - 1);
  float v = (r < G3 && k < HS_) ? w_hh[r * HS_ + k] : 0.0f;
  whh_b[idx] = f2bf(v);
}

__global__ __launch_bounds__(256) void prep_gm(const float* __restrict__ gate_w,
                                               const float* __restrict__ map_w,
                                               unsigned short* __restrict__ gm_b)
{
  int idx = blockIdx.x * 256 + threadIdx.x;       // GMN * HSP
  if (idx >= GMN * HSP) return;
  int r = idx >> 9, k = idx & (HSP - 1);
  float v = 0.0f;
  if (k < HS_) {
    if (r < HS_)                        v = gate_w[r * VS_ + k];          // gate rows 0..500
    else if (r >= 512 && r < 512 + HS_) v = map_w[(r - 512) * VS_ + k];   // map rows 512..1012
  }
  gm_b[idx] = f2bf(v);
}

__global__ __launch_bounds__(256) void zero_f32(float* __restrict__ p, int n)
{
  int idx = blockIdx.x * 256 + threadIdx.x;
  if (idx < n) p[idx] = 0.0f;
}

// ---------------------------------------------------------------------------
// h_in[b,:] = sum_{n<v} adj[b,v,n] * M[b,n,:]   (write fp32 + bf16, 512-padded)
// ---------------------------------------------------------------------------
__global__ __launch_bounds__(256) void hin_kernel(const float* __restrict__ Mst,
                                                  const float* __restrict__ adj,
                                                  float* __restrict__ hin_f,
                                                  unsigned short* __restrict__ hin_b,
                                                  int v)
{
  const int b = blockIdx.x;
  const int j = threadIdx.x;                       // handles j and j+256
  const float* Mb = Mst + (size_t)b * MAXN * HSP;
  const float* ab = adj + (size_t)b * (MAXN * MAXN) + v * MAXN;
  float s0 = 0.0f, s1 = 0.0f;
  for (int n = 0; n < v; ++n) {
    float c = ab[n];
    s0 += c * Mb[n * HSP + j];
    s1 += c * Mb[n * HSP + j + 256];
  }
  size_t o = (size_t)b * HSP;
  hin_f[o + j]       = s0;
  hin_f[o + j + 256] = s1;
  hin_b[o + j]       = f2bf(s0);
  hin_b[o + j + 256] = f2bf(s1);
}

// ---------------------------------------------------------------------------
// GRU elementwise: hv = (1-z)*n + z*h_in, written fp32 + bf16 (512-padded)
// ---------------------------------------------------------------------------
__global__ __launch_bounds__(256) void gru_kernel(const float* __restrict__ gh,
                                                  const int* __restrict__ node_types,
                                                  const float* __restrict__ w_ih,
                                                  const float* __restrict__ b_ih,
                                                  const float* __restrict__ b_hh,
                                                  const float* __restrict__ hin_f,
                                                  float* __restrict__ hv_f,
                                                  unsigned short* __restrict__ hv_b,
                                                  int v)
{
  int idx = blockIdx.x * 256 + threadIdx.x;        // B_ * HSP
  if (idx >= B_ * HSP) return;
  int b = idx >> 9, j = idx & (HSP - 1);
  if (j >= HS_) { hv_f[idx] = 0.0f; hv_b[idx] = 0; return; }

  const int t = node_types[b * MAXN + v];
  const float* ghb = gh + (size_t)b * G3P;
  float hr = ghb[j]            + b_hh[j];
  float hz = ghb[j + HS_]      + b_hh[j + HS_];
  float hn = ghb[j + 2 * HS_]  + b_hh[j + 2 * HS_];
  float ir = w_ih[(j)           * NVT_ + t] + b_ih[j];
  float iz = w_ih[(j + HS_)     * NVT_ + t] + b_ih[j + HS_];
  float in = w_ih[(j + 2 * HS_) * NVT_ + t] + b_ih[j + 2 * HS_];

  float r = sigmoidf_(ir + hr);
  float z = sigmoidf_(iz + hz);
  float n = tanhf(in + r * hn);
  float h = hin_f[idx];
  float hv = (1.0f - z) * n + z * h;
  hv_f[idx] = hv;
  hv_b[idx] = f2bf(hv);
}

// ---------------------------------------------------------------------------
// M[b,v,j] = sigmoid(gm_gate + gate_w[j,HS+v] + gate_b) * (gm_map + map_w[j,HS+v])
// ---------------------------------------------------------------------------
__global__ __launch_bounds__(256) void gatemap_kernel(const float* __restrict__ gm,
                                                      const float* __restrict__ gate_w,
                                                      const float* __restrict__ gate_b,
                                                      const float* __restrict__ map_w,
                                                      float* __restrict__ Mst,
                                                      int v)
{
  int idx = blockIdx.x * 256 + threadIdx.x;        // B_ * HSP
  if (idx >= B_ * HSP) return;
  int b = idx >> 9, j = idx & (HSP - 1);
  float out = 0.0f;
  if (j < HS_) {
    const float* gmb = gm + (size_t)b * GMN;
    float g = sigmoidf_(gmb[j] + gate_w[j * VS_ + HS_ + v] + gate_b[j]);
    float m = gmb[512 + j] + map_w[j * VS_ + HS_ + v];
    out = g * m;
  }
  Mst[(size_t)b * MAXN * HSP + (size_t)v * HSP + j] = out;
}

// ---------------------------------------------------------------------------
// Final FC: out[b, 0:56] = hg @ fc1_w.T + fc1_b ; out[b, 56:112] = fc2
// ---------------------------------------------------------------------------
__global__ __launch_bounds__(256) void fc_kernel(const float* __restrict__ hg,
                                                 const float* __restrict__ fc1_w,
                                                 const float* __restrict__ fc1_b,
                                                 const float* __restrict__ fc2_w,
                                                 const float* __restrict__ fc2_b,
                                                 float* __restrict__ out)
{
  int idx = blockIdx.x * 256 + threadIdx.x;        // B_ * 2*NZ_
  if (idx >= B_ * 2 * NZ_) return;
  int b = idx / (2 * NZ_);
  int o = idx - b * (2 * NZ_);
  const float* w;
  float s;
  if (o < NZ_) { w = fc1_w + (size_t)o * HS_;         s = fc1_b[o]; }
  else         { w = fc2_w + (size_t)(o - NZ_) * HS_; s = fc2_b[o - NZ_]; }
  const float* h = hg + (size_t)b * HSP;
  for (int k = 0; k < HS_; ++k) s += w[k] * h[k];
  out[idx] = s;
}

// ---------------------------------------------------------------------------
extern "C" void kernel_launch(void* const* d_in, const int* in_sizes, int n_in,
                              void* d_out, int out_size, void* d_ws, size_t ws_size,
                              hipStream_t stream) {
  (void)in_sizes; (void)n_in; (void)out_size; (void)ws_size;

  const int*   node_types = (const int*)  d_in[0];
  const float* adj        = (const float*)d_in[1];
  const float* w_ih       = (const float*)d_in[2];
  const float* w_hh       = (const float*)d_in[3];
  const float* b_ih       = (const float*)d_in[4];
  const float* b_hh       = (const float*)d_in[5];
  const float* gate_w     = (const float*)d_in[6];
  const float* gate_b     = (const float*)d_in[7];
  const float* map_w      = (const float*)d_in[8];
  const float* fc1_w      = (const float*)d_in[9];
  const float* fc1_b      = (const float*)d_in[10];
  const float* fc2_w      = (const float*)d_in[11];
  const float* fc2_b      = (const float*)d_in[12];
  float* out = (float*)d_out;

  // Workspace carve-up (256B aligned)
  char*  ws  = (char*)d_ws;
  size_t off = 0;
  auto carve = [&](size_t bytes) -> void* {
    void* p = ws + off;
    off += bytes;
    off = (off + 255) & ~(size_t)255;
    return p;
  };
  float*          Mst   = (float*)         carve((size_t)B_ * MAXN * HSP * 4); // 64 MB
  float*          gh    = (float*)         carve((size_t)B_ * G3P * 4);        // 12 MB
  float*          gm    = (float*)         carve((size_t)B_ * GMN * 4);        // 8 MB
  float*          hin_f = (float*)         carve((size_t)B_ * HSP * 4);
  unsigned short* hin_b = (unsigned short*)carve((size_t)B_ * HSP * 2);
  float*          hv_f  = (float*)         carve((size_t)B_ * HSP * 4);
  unsigned short* hv_b  = (unsigned short*)carve((size_t)B_ * HSP * 2);
  unsigned short* whh_b = (unsigned short*)carve((size_t)G3P * HSP * 2);
  unsigned short* gmw_b = (unsigned short*)carve((size_t)GMN * HSP * 2);

  // Preprocess: zero M, bf16-convert weights
  {
    int n = B_ * MAXN * HSP;
    zero_f32<<<(n + 255) / 256, 256, 0, stream>>>(Mst, n);
  }
  prep_whh<<<(G3P * HSP + 255) / 256, 256, 0, stream>>>(w_hh, whh_b);
  prep_gm <<<(GMN * HSP + 255) / 256, 256, 0, stream>>>(gate_w, map_w, gmw_b);

  const dim3 blk(256);
  const dim3 grid_gemm1(G3P / 256, B_ / 64);   //  6 x 32
  const dim3 grid_gemm2(GMN / 256, B_ / 64);   //  4 x 32
  const int  nElem = B_ * HSP;                 // 1,048,576
  const int  gElem = nElem / 256;              // 4096 blocks

  for (int v = 0; v < MAXN; ++v) {
    hin_kernel<<<B_, blk, 0, stream>>>(Mst, adj, hin_f, hin_b, v);
    wmma_gemm_bf16<<<grid_gemm1, blk, 0, stream>>>(hin_b, whh_b, gh, G3P);
    gru_kernel<<<gElem, blk, 0, stream>>>(gh, node_types, w_ih, b_ih, b_hh,
                                          hin_f, hv_f, hv_b, v);
    wmma_gemm_bf16<<<grid_gemm2, blk, 0, stream>>>(hv_b, gmw_b, gm, GMN);
    gatemap_kernel<<<gElem, blk, 0, stream>>>(gm, gate_w, gate_b, map_w, Mst, v);
  }

  // hv_f now holds hg (step 15 hidden state)
  fc_kernel<<<(B_ * 2 * NZ_ + 255) / 256, blk, 0, stream>>>(hv_f, fc1_w, fc1_b,
                                                            fc2_w, fc2_b, out);
}
